// PNATower_62225486185135
// MI455X (gfx1250) — compile-verified, matrix operand order checked by reading, not run on hardware
//
#include <hip/hip_runtime.h>

// ---------------------------------------------------------------------------
// PNA tower forward for MI455X (gfx1250, wave32, WMMA bf16 16x16x32).
//   h:[50000,128] e:[800000,128] snorm:[50000,1] src/dst:[800000] (i32)
//   W_pre:[384,128] b_pre:[128] W_post:[1664,128] b_post:[128]
//   out:[50000,128] f32
// ---------------------------------------------------------------------------

#define D        128
#define NNODES   50000
#define NEDGES   800000
#define ETILES   (NEDGES / 16)   // 50000, exact
#define NTILES   (NNODES / 16)   // 3125,  exact
#define KBPRE    12              // 384 / 32
#define KBPOST   52              // 1664 / 32
#define AVG_D_LOG 2.833f
#define EPSV      1e-5f

// workspace byte offsets
#define OFF_PPRE   0u                       // 12*8*32*16 bf16  = 98304 B
#define OFF_PPOST  98304u                   // 52*8*32*16 bf16  = 425984 B
#define OFF_SUM    524288u
#define SZ_NODEF   25600000u                // 50000*128*4
#define OFF_SQ     (OFF_SUM + SZ_NODEF)
#define OFF_MX     (OFF_SQ  + SZ_NODEF)
#define OFF_MN     (OFF_MX  + SZ_NODEF)
#define OFF_DEG    (OFF_MN  + SZ_NODEF)     // 50000*4 B   -> ends 103124288
#define OFF_HBF    103124288u               // 50000*128*2 = 12.8 MB bf16 h

typedef __attribute__((ext_vector_type(16))) __bf16 v16bf;
typedef __attribute__((ext_vector_type(8)))  float  v8f;

// Matches the async-LDS builtin's parameter type per hipcc diagnostic:
// "__attribute__((__vector_size__(4 * sizeof(int)))) int *"
typedef int vsi4 __attribute__((__vector_size__(4 * sizeof(int))));
typedef __attribute__((address_space(1))) vsi4* global_vsi4_ptr;
typedef __attribute__((address_space(3))) vsi4* lds_vsi4_ptr;

union Frag16 {
  v16bf v;
  uint4 q[2];
  unsigned u[8];
};

// f32 pair -> packed bf16 dword (round half up). v_perm_b32 gathers the two
// high halves in one op: 3 VALU total, no 16-bit mov shuffles.
#if __has_builtin(__builtin_amdgcn_perm)
__device__ __forceinline__ unsigned pack2bf(float lo, float hi) {
  return __builtin_amdgcn_perm(__float_as_uint(hi) + 0x8000u,
                               __float_as_uint(lo) + 0x8000u, 0x07060302u);
}
#else
__device__ __forceinline__ unsigned pack2bf(float lo, float hi) {
  unsigned a = __float_as_uint(lo);
  unsigned b = __float_as_uint(hi);
  a = (a + 0x7fffu + ((a >> 16) & 1u)) >> 16;
  b = (b + 0x7fffu + ((b >> 16) & 1u)) >> 16;
  return a | (b << 16);
}
#endif

// float atomic max/min via sign-aware integer atomics (single L2 atomic op)
__device__ __forceinline__ void atomicMaxF(float* a, float v) {
  if (v >= 0.0f) atomicMax((int*)a, __float_as_int(v));
  else           atomicMin((unsigned int*)a, __float_as_uint(v));
}
__device__ __forceinline__ void atomicMinF(float* a, float v) {
  if (v >= 0.0f) atomicMin((int*)a, __float_as_int(v));
  else           atomicMax((unsigned int*)a, __float_as_uint(v));
}

// ---------------------------------------------------------------------------
// Kernel 1: pack fp32 weights into bf16 WMMA B-fragment layout.
// Fragment id = (kb*8 + nb)*32 + lane; per lane 16 bf16 (32B) contiguous.
// Per-lane K map (16-bit operand layout): K = kb*32 + (lane>=16 ? 8 : 0)
//                                             + (t < 8 ? t : t + 8)
// ---------------------------------------------------------------------------
__global__ void pack_weights(const float* __restrict__ Wpre,
                             const float* __restrict__ Wpost,
                             unsigned* __restrict__ Ppre,
                             unsigned* __restrict__ Ppost) {
  int id = blockIdx.x * blockDim.x + threadIdx.x;
  const int NF_PRE  = KBPRE  * 8 * 32;   // 3072
  const int NF_POST = KBPOST * 8 * 32;   // 13312
  if (id >= NF_PRE + NF_POST) return;
  const float* W;
  unsigned* P;
  int frag;
  if (id < NF_PRE) { W = Wpre;  P = Ppre;  frag = id; }
  else             { W = Wpost; P = Ppost; frag = id - NF_PRE; }
  int lane = frag & 31;
  int nb   = (frag >> 5) & 7;
  int kb   = frag >> 8;
  int n     = nb * 16 + (lane & 15);
  int kbase = kb * 32 + (lane >> 4) * 8;
  unsigned u[8];
#pragma unroll
  for (int t2 = 0; t2 < 8; t2++) {
    int t0 = 2 * t2, t1 = t0 + 1;
    int k0 = kbase + (t0 < 8 ? t0 : t0 + 8);
    int k1 = kbase + (t1 < 8 ? t1 : t1 + 8);
    u[t2] = pack2bf(W[(size_t)k0 * D + n], W[(size_t)k1 * D + n]);
  }
  uint4* o = (uint4*)(P + (size_t)frag * 8);
  o[0] = make_uint4(u[0], u[1], u[2], u[3]);
  o[1] = make_uint4(u[4], u[5], u[6], u[7]);
}

// ---------------------------------------------------------------------------
// Kernel 2a: convert h to a row-major bf16 copy (A operands load directly).
// ---------------------------------------------------------------------------
__global__ void conv_h_bf16(const float* __restrict__ h,
                            unsigned* __restrict__ Hbf) {
  int i = blockIdx.x * blockDim.x + threadIdx.x;
  if (i >= NNODES * (D / 2)) return;
  float2 p = ((const float2*)h)[i];
  Hbf[i] = pack2bf(p.x, p.y);
}

// ---------------------------------------------------------------------------
// Kernel 2b: init mailboxes (ws is poisoned before timing; must re-init).
// ---------------------------------------------------------------------------
__global__ void init_mailboxes(float* __restrict__ sum, float* __restrict__ sq,
                               float* __restrict__ mx, float* __restrict__ mn,
                               float* __restrict__ deg) {
  int i = blockIdx.x * blockDim.x + threadIdx.x;
  if (i < NNODES * D) {
    sum[i] = 0.0f;
    sq[i]  = 0.0f;
    mx[i]  = __uint_as_float(0xff800000u);  // -inf
    mn[i]  = __uint_as_float(0x7f800000u);  // +inf
  }
  if (i < NNODES) deg[i] = 0.0f;
}

// ---------------------------------------------------------------------------
// Kernel 3: fused edge GEMM + multi-aggregator scatter.
// One wave32 per 16-edge tile. A rows = [h[src] | h[dst] | e]; h segments
// load pre-converted bf16 directly (2 x b128, zero VALU), e converts with
// v_perm packing. W_pre (bf16, 96KB) staged in LDS via CDNA5 async DMA.
// ---------------------------------------------------------------------------
__global__ __launch_bounds__(256, 1) void edge_gemm_scatter(
    const float* __restrict__ e,
    const unsigned short* __restrict__ Hbf,
    const int* __restrict__ src, const int* __restrict__ dst,
    const unsigned* __restrict__ Ppre, const float* __restrict__ b_pre,
    float* __restrict__ sum, float* __restrict__ sq,
    float* __restrict__ mx, float* __restrict__ mn,
    float* __restrict__ deg) {
  __shared__ __align__(16) unsigned ldsB[KBPRE * 8 * 32 * 8];  // 96 KB

#if __has_builtin(__builtin_amdgcn_global_load_async_to_lds_b128)
  // Direct memory->LDS DMA, no VGPR round trip; tracked by ASYNCcnt.
  for (int c = threadIdx.x; c < KBPRE * 8 * 32 * 2; c += 256) {  // 6144 x b128
    __builtin_amdgcn_global_load_async_to_lds_b128(
        (global_vsi4_ptr)(Ppre + (size_t)c * 4),
        (lds_vsi4_ptr)&ldsB[c * 4], 0, 0);
  }
#if __has_builtin(__builtin_amdgcn_s_wait_asynccnt)
  __builtin_amdgcn_s_wait_asynccnt(0);
#else
  asm volatile("s_wait_asynccnt 0x0" ::: "memory");
#endif
#else
  for (int i = threadIdx.x; i < KBPRE * 8 * 32 * 8; i += 256) ldsB[i] = Ppre[i];
#endif
  __syncthreads();

  const int lane = threadIdx.x & 31;
  const int wave = threadIdx.x >> 5;
  const int mrow = lane & 15;
  const int half = lane >> 4;

  float bp[8];
#pragma unroll
  for (int nb = 0; nb < 8; nb++) bp[nb] = b_pre[nb * 16 + mrow];

  for (int tile = blockIdx.x * 8 + wave; tile < ETILES; tile += gridDim.x * 8) {
    const int i = tile * 16 + mrow;  // edge handled by this lane's A row
    const unsigned short* rsb = Hbf + (size_t)src[i] * D + half * 8;
    const unsigned short* rdb = Hbf + (size_t)dst[i] * D + half * 8;
    const float*          re  = e   + (size_t)i * D + half * 8;

    v8f acc[8] = {};
#pragma unroll
    for (int kb = 0; kb < KBPRE; kb++) {
      const int kbase = kb * 32;
      Frag16 A;
      if (kb < 8) {
        // h[src] / h[dst]: fragment runs are contiguous bf16 in Hbf
        const unsigned short* rowb =
            (kb < 4) ? (rsb + kbase) : (rdb + (kbase - 128));
        A.q[0] = *(const uint4*)(rowb);
        A.q[1] = *(const uint4*)(rowb + 16);
      } else {
        // e: fp32 from HBM (read exactly once), convert in-register
        const float* row = re + (kbase - 256);
        float4 f0 = *(const float4*)(row);
        float4 f1 = *(const float4*)(row + 4);
        float4 f2 = *(const float4*)(row + 16);
        float4 f3 = *(const float4*)(row + 20);
        A.u[0] = pack2bf(f0.x, f0.y); A.u[1] = pack2bf(f0.z, f0.w);
        A.u[2] = pack2bf(f1.x, f1.y); A.u[3] = pack2bf(f1.z, f1.w);
        A.u[4] = pack2bf(f2.x, f2.y); A.u[5] = pack2bf(f2.z, f2.w);
        A.u[6] = pack2bf(f3.x, f3.y); A.u[7] = pack2bf(f3.z, f3.w);
      }
#pragma unroll
      for (int nb = 0; nb < 8; nb++) {
        Frag16 B;
        const uint4* p = (const uint4*)&ldsB[((kb * 8 + nb) * 32 + lane) * 8];
        B.q[0] = p[0];
        B.q[1] = p[1];
        acc[nb] = __builtin_amdgcn_wmma_f32_16x16x32_bf16(
            false, A.v, false, B.v, (short)0, acc[nb], false, false);
      }
    }

    // scatter epilogue: C/D layout row M = r + 8*half, col = nb*16 + mrow
#pragma unroll
    for (int r = 0; r < 8; r++) {
      const int ic   = tile * 16 + r + half * 8;
      const int node = dst[ic];
      float* srow = sum + (size_t)node * D;
      float* qrow = sq  + (size_t)node * D;
      float* xrow = mx  + (size_t)node * D;
      float* nrow = mn  + (size_t)node * D;
      if (mrow == 0) atomicAdd(deg + node, 1.0f);  // once per edge
#pragma unroll
      for (int nb = 0; nb < 8; nb++) {
        const int col = nb * 16 + mrow;
        const float v = acc[nb][r] + bp[nb];
        atomicAdd(srow + col, v);
        atomicAdd(qrow + col, v * v);
        atomicMaxF(xrow + col, v);
        atomicMinF(nrow + col, v);
      }
    }
  }
}

// ---------------------------------------------------------------------------
// Kernel 4 helper: synthesize 8 consecutive elements of the virtual
// aggregated feature row ({mean,max,min,std} x {id,amp,att}), seg >= 1.
// ---------------------------------------------------------------------------
__device__ __forceinline__ void node_run8(
    float* o, int seg, int j,
    const float* __restrict__ srow, const float* __restrict__ qrow,
    const float* __restrict__ xrow, const float* __restrict__ nrow,
    float invd, bool has, float ampf, float attf) {
  const int blk  = seg - 1;     // 0..11
  const int kind = blk & 3;     // mean / max / min / std
  const float scl = (blk < 4) ? 1.0f : (blk < 8) ? ampf : attf;
  float vals[8];
  if (kind == 0) {
    float4 s0 = *(const float4*)(srow + j), s1 = *(const float4*)(srow + j + 4);
    float sv[8] = {s0.x, s0.y, s0.z, s0.w, s1.x, s1.y, s1.z, s1.w};
#pragma unroll
    for (int t = 0; t < 8; t++) vals[t] = sv[t] * invd;
  } else if (kind == 1) {
    float4 x0 = *(const float4*)(xrow + j), x1 = *(const float4*)(xrow + j + 4);
    float xv[8] = {x0.x, x0.y, x0.z, x0.w, x1.x, x1.y, x1.z, x1.w};
#pragma unroll
    for (int t = 0; t < 8; t++) vals[t] = xv[t];
  } else if (kind == 2) {
    float4 n0 = *(const float4*)(nrow + j), n1 = *(const float4*)(nrow + j + 4);
    float nv[8] = {n0.x, n0.y, n0.z, n0.w, n1.x, n1.y, n1.z, n1.w};
#pragma unroll
    for (int t = 0; t < 8; t++) vals[t] = nv[t];
  } else {
    float4 s0 = *(const float4*)(srow + j), s1 = *(const float4*)(srow + j + 4);
    float4 q0 = *(const float4*)(qrow + j), q1 = *(const float4*)(qrow + j + 4);
    float sv[8] = {s0.x, s0.y, s0.z, s0.w, s1.x, s1.y, s1.z, s1.w};
    float qv[8] = {q0.x, q0.y, q0.z, q0.w, q1.x, q1.y, q1.z, q1.w};
#pragma unroll
    for (int t = 0; t < 8; t++) {
      float mean = sv[t] * invd;
      float var  = fmaxf(qv[t] * invd - mean * mean, 0.0f);
      vals[t] = __builtin_sqrtf(var + EPSV);
    }
  }
#pragma unroll
  for (int t = 0; t < 8; t++) o[t] = has ? vals[t] * scl : 0.0f;
}

// ---------------------------------------------------------------------------
// Kernel 4: fused node post-processing + post GEMM + graph_norm.
// One wave32 per 16-node tile; [N,1664] feature matrix never materialized.
// ---------------------------------------------------------------------------
__global__ __launch_bounds__(256, 1) void node_gemm(
    const unsigned short* __restrict__ Hbf, const float* __restrict__ snorm,
    const unsigned* __restrict__ Ppost, const float* __restrict__ b_post,
    const float* __restrict__ sum, const float* __restrict__ sq,
    const float* __restrict__ mx, const float* __restrict__ mn,
    const float* __restrict__ deg, float* __restrict__ out) {
  const int lane = threadIdx.x & 31;
  const int wave = threadIdx.x >> 5;
  const int tile = blockIdx.x * 8 + wave;
  if (tile >= NTILES) return;  // wave-uniform; EXEC stays full for WMMA
  const int mrow = lane & 15;
  const int half = lane >> 4;

  const int na = tile * 16 + mrow;  // node feeding this lane's A row
  const unsigned short* hrowb = Hbf + (size_t)na * D;
  const float* srow = sum + (size_t)na * D;
  const float* qrow = sq  + (size_t)na * D;
  const float* xrow = mx  + (size_t)na * D;
  const float* nrow = mn  + (size_t)na * D;

  const float dg   = deg[na];
  const bool  has  = dg > 0.0f;
  const float dgs  = fmaxf(dg, 1.0f);
  const float invd = 1.0f / dgs;
  const float ampf = __builtin_logf(dg + 1.0f) * (1.0f / AVG_D_LOG);
  const float attf = AVG_D_LOG / __builtin_logf(dgs + 1.0f);

  v8f acc[8] = {};
  for (int kb = 0; kb < KBPOST; kb++) {
    const int kbase = kb * 32;
    const int seg   = kbase >> 7;                 // 0..12 (uniform per kb)
    const int j0    = (kbase & 127) + half * 8;
    Frag16 A;
    if (seg == 0) {
      A.q[0] = *(const uint4*)(hrowb + j0);
      A.q[1] = *(const uint4*)(hrowb + j0 + 16);
    } else {
      float a[16];
      node_run8(a,     seg, j0,      srow, qrow, xrow, nrow, invd, has, ampf, attf);
      node_run8(a + 8, seg, j0 + 16, srow, qrow, xrow, nrow, invd, has, ampf, attf);
#pragma unroll
      for (int t2 = 0; t2 < 8; t2++) A.u[t2] = pack2bf(a[2 * t2], a[2 * t2 + 1]);
    }
#pragma unroll
    for (int nb = 0; nb < 8; nb++) {
      Frag16 B;
      const uint4* p = (const uint4*)(Ppost + ((size_t)(kb * 8 + nb) * 32 + lane) * 8);
      B.q[0] = p[0];
      B.q[1] = p[1];
      acc[nb] = __builtin_amdgcn_wmma_f32_16x16x32_bf16(
          false, A.v, false, B.v, (short)0, acc[nb], false, false);
    }
  }

  float bp[8];
#pragma unroll
  for (int nb = 0; nb < 8; nb++) bp[nb] = b_post[nb * 16 + mrow];
#pragma unroll
  for (int r = 0; r < 8; r++) {
    const int nc = tile * 16 + r + half * 8;  // C/D row
    const float sn = snorm[nc];
    float* orow = out + (size_t)nc * D;
#pragma unroll
    for (int nb = 0; nb < 8; nb++) {
      orow[nb * 16 + mrow] = (acc[nb][r] + bp[nb]) * sn;
    }
  }
}

// ---------------------------------------------------------------------------
extern "C" void kernel_launch(void* const* d_in, const int* in_sizes, int n_in,
                              void* d_out, int out_size, void* d_ws, size_t ws_size,
                              hipStream_t stream) {
  const float* h      = (const float*)d_in[0];
  const float* e      = (const float*)d_in[1];
  const float* snorm  = (const float*)d_in[2];
  const int*   src    = (const int*)d_in[3];
  const int*   dst    = (const int*)d_in[4];
  const float* Wpre   = (const float*)d_in[5];
  const float* bpre   = (const float*)d_in[6];
  const float* Wpost  = (const float*)d_in[7];
  const float* bpost  = (const float*)d_in[8];
  float* out = (float*)d_out;

  char* ws = (char*)d_ws;
  unsigned* Ppre  = (unsigned*)(ws + OFF_PPRE);
  unsigned* Ppost = (unsigned*)(ws + OFF_PPOST);
  float* sum = (float*)(ws + OFF_SUM);
  float* sq  = (float*)(ws + OFF_SQ);
  float* mx  = (float*)(ws + OFF_MX);
  float* mn  = (float*)(ws + OFF_MN);
  float* deg = (float*)(ws + OFF_DEG);
  unsigned*       HbfU = (unsigned*)(ws + OFF_HBF);
  unsigned short* Hbf  = (unsigned short*)(ws + OFF_HBF);

  pack_weights<<<(16384 + 255) / 256, 256, 0, stream>>>(Wpre, Wpost, Ppre, Ppost);
  conv_h_bf16<<<(NNODES * (D / 2) + 255) / 256, 256, 0, stream>>>(h, HbfU);
  init_mailboxes<<<(NNODES * D + 255) / 256, 256, 0, stream>>>(sum, sq, mx, mn, deg);
  edge_gemm_scatter<<<ETILES / 8, 256, 0, stream>>>(e, Hbf, src, dst, Ppre, bpre,
                                                    sum, sq, mx, mn, deg);
  node_gemm<<<(NTILES + 7) / 8, 256, 0, stream>>>(Hbf, snorm, Ppost, bpost,
                                                  sum, sq, mx, mn, deg, out);
}
